// AttnBlock_4767413699312
// MI455X (gfx1250) — matrix operand order repaired and checked by reading, not compiled, not run on hardware
//
#include <hip/hip_runtime.h>

typedef _Float16 half_t;
typedef __attribute__((ext_vector_type(8)))  _Float16 v8h;
typedef __attribute__((ext_vector_type(16))) _Float16 v16h;
typedef __attribute__((ext_vector_type(8)))  float    v8f;

#define RTOT  32768   // B * H * W tokens
#define CDIM  512
#define FDIM  1536
#define NTOK  1024
#define NH    8
#define DH    64

__device__ __forceinline__ float wsum(float v){
#pragma unroll
  for (int m = 16; m >= 1; m >>= 1) v += __shfl_xor(v, m, 32);
  return v;
}
__device__ __forceinline__ float wmaxr(float v){
#pragma unroll
  for (int m = 16; m >= 1; m >>= 1) v = fmaxf(v, __shfl_xor(v, m, 32));
  return v;
}

// A fragment (16x32 f16): this lane's row pointer at k=0 of the 32-wide tile.
// lanes 0-15: K = {khalf..khalf+7, khalf+16..khalf+23} with khalf=0; lanes 16-31: khalf=8.
__device__ __forceinline__ v16h load_a_frag(const half_t* rowp, int khalf){
  v16h r;
  ((v8h*)&r)[0] = *(const v8h*)(rowp + khalf);
  ((v8h*)&r)[1] = *(const v8h*)(rowp + khalf + 16);
  return r;
}
// B fragment (32x16 f16): this lane's column pointer (K-contiguous storage).
// lanes 0-15 hold K={0..7,16..23}, lanes 16-31 hold K={8..15,24..31} (kb = 0 or 16).
__device__ __forceinline__ v16h load_b_frag(const half_t* colp, int kb){
  v16h r;
  ((v8h*)&r)[0] = *(const v8h*)(colp + kb / 2);
  ((v8h*)&r)[1] = *(const v8h*)(colp + kb / 2 + 16);
  return r;
}
#define WMMA(a,b,c) __builtin_amdgcn_wmma_f32_16x16x32_f16(false,(a),false,(b),(short)0,(c),false,false)

// ---------------- Kernel 0: weight transpose + f32->f16 convert --------------------
// W [512 x ncols] f32 row-major  ->  wT [ncols x 512] f16 (B-fragment layout).
__global__ __launch_bounds__(256) void k_wprep(const float* __restrict__ W,
                                               int ncols, half_t* __restrict__ wT){
  const int n = blockIdx.x;
  for (int k = threadIdx.x; k < CDIM; k += 256)
    wT[(size_t)n * CDIM + k] = (half_t)W[(size_t)k * ncols + n];
}

// ---------------- Kernel 1: LayerNorm1 -> f16 ----------------
__global__ __launch_bounds__(256) void k_ln1(const float* __restrict__ x,
                                             const float* __restrict__ g,
                                             half_t* __restrict__ xn){
  const int lane = threadIdx.x & 31, wid = threadIdx.x >> 5;
  const size_t row = (size_t)blockIdx.x * 8 + wid;
  const float* xp = x + row * CDIM + lane * 16;
  float v[16]; float s = 0.f, sq = 0.f;
#pragma unroll
  for (int i = 0; i < 16; i += 4){
    float4 t = *(const float4*)(xp + i);
    v[i] = t.x; v[i+1] = t.y; v[i+2] = t.z; v[i+3] = t.w;
  }
#pragma unroll
  for (int i = 0; i < 16; i++){ s += v[i]; sq += v[i]*v[i]; }
  s = wsum(s); sq = wsum(sq);
  const float mu = s * (1.f/CDIM);
  const float var = sq * (1.f/CDIM) - mu*mu;
  const float rs = rsqrtf(var + 1e-5f);
  const float* gp = g + lane * 16;
  v8h o0, o1;
#pragma unroll
  for (int i = 0; i < 8; i++)  o0[i] = (half_t)((v[i]   - mu)*rs*gp[i]);
#pragma unroll
  for (int i = 0; i < 8; i++)  o1[i] = (half_t)((v[i+8] - mu)*rs*gp[i+8]);
  *(v8h*)(xn + row*CDIM + lane*16)     = o0;
  *(v8h*)(xn + row*CDIM + lane*16 + 8) = o1;
}

// ---------------- GEMM: [RTOT x 512](f16) x wT[wcols x 512](f16) ------------------
// Block: 256 thr (8 waves), tile 64(M) x 128(N), waves 2x4, each 2x2 WMMA tiles.
// No LDS: A and B fragments are direct 16B global loads; wT is L2-resident.
template<bool HALF_OUT>
__global__ __launch_bounds__(256) void k_gemm(const half_t* __restrict__ A,
                                              const half_t* __restrict__ WT,
                                              int wcols, void* __restrict__ outp){
  const int tid = threadIdx.x;
  const int lane = tid & 31, wid = tid >> 5;
  const int l16 = lane & 15, hi = lane >> 4;
  const int khalf = hi * 8, kb = hi * 16;
  const int colBase = blockIdx.x * 128;
  const int rowBase = blockIdx.y * 64;
  const int mw = wid >> 2, nw = wid & 3;
  const int rm  = rowBase + mw * 32;
  const int cn0 = colBase + nw * 32;
  v8f c00 = {}, c01 = {}, c10 = {}, c11 = {};
  const half_t* a0p = A  + (size_t)(rm + l16) * CDIM;
  const half_t* a1p = a0p + 16 * CDIM;
  const half_t* b0p = WT + (size_t)(cn0 + l16) * CDIM;
  const half_t* b1p = b0p + 16 * CDIM;
  for (int k0 = 0; k0 < CDIM; k0 += 32){
    v16h a0 = load_a_frag(a0p + k0, khalf);
    v16h a1 = load_a_frag(a1p + k0, khalf);
    v16h b0 = load_b_frag(b0p + k0, kb);
    v16h b1 = load_b_frag(b1p + k0, kb);
    c00 = WMMA(a0, b0, c00);
    c01 = WMMA(a0, b1, c01);
    c10 = WMMA(a1, b0, c10);
    c11 = WMMA(a1, b1, c11);
  }
  const int mrow = rm + hi * 8;
  const size_t colg = (size_t)cn0 + l16;
#pragma unroll
  for (int v = 0; v < 8; ++v){
    const size_t r0 = (size_t)(mrow + v) * wcols, r1 = (size_t)(mrow + 16 + v) * wcols;
    if (HALF_OUT){
      half_t* o = (half_t*)outp;
      o[r0 + colg]      = (half_t)c00[v]; o[r0 + colg + 16] = (half_t)c01[v];
      o[r1 + colg]      = (half_t)c10[v]; o[r1 + colg + 16] = (half_t)c11[v];
    } else {
      float* o = (float*)outp;
      o[r0 + colg]      = c00[v]; o[r0 + colg + 16] = c01[v];
      o[r1 + colg]      = c10[v]; o[r1 + colg + 16] = c11[v];
    }
  }
}

// ---------------- Kernel 3: q softmax over dim_head, /sqrt(64); write [b,h,n,d] ----
__global__ __launch_bounds__(256) void k_qsm(const half_t* __restrict__ qkv,
                                             half_t* __restrict__ qT){
  const int lane = threadIdx.x & 31, wid = threadIdx.x >> 5;
  const size_t pair = (size_t)blockIdx.x * 8 + wid;  // (row, head)
  const size_t row = pair >> 3;
  const int h = (int)(pair & 7);
  const half_t* p = qkv + row * FDIM + h * DH + lane * 2;
  const float v0 = (float)p[0], v1 = (float)p[1];
  const float M = wmaxr(fmaxf(v0, v1));
  const float e0 = __expf(v0 - M), e1 = __expf(v1 - M);
  const float s = wsum(e0 + e1);
  const float inv = 1.f / (s * 8.f);                 // softmax then /sqrt(DH)
  const size_t b = row >> 10, n = row & 1023;
  half_t* o = qT + ((b * NH + h) * NTOK + n) * DH + lane * 2;
  o[0] = (half_t)(e0 * inv); o[1] = (half_t)(e1 * inv);
}

// ---------------- Kernel 4: k softmax over tokens; v/N; both written [b,h,d,n] -----
__global__ __launch_bounds__(256) void k_ksm_v(const half_t* __restrict__ qkv,
                                               half_t* __restrict__ kT,
                                               half_t* __restrict__ vT){
  __shared__ float red[4][64];
  const int bh = blockIdx.x;
  const size_t b = (size_t)(bh >> 3);
  const int h = bh & 7;
  const int d = threadIdx.x & 63, cg = threadIdx.x >> 6;
  const half_t* kbase = qkv + b * NTOK * (size_t)FDIM + CDIM + h * DH + d;
  float pm = -1e30f;
  for (int n = cg; n < NTOK; n += 4)
    pm = fmaxf(pm, (float)kbase[(size_t)n * FDIM]);
  red[cg][d] = pm; __syncthreads();
  const float M = fmaxf(fmaxf(red[0][d], red[1][d]), fmaxf(red[2][d], red[3][d]));
  __syncthreads();
  float ps = 0.f;
  for (int n = cg; n < NTOK; n += 4)
    ps += __expf((float)kbase[(size_t)n * FDIM] - M);
  red[cg][d] = ps; __syncthreads();
  const float S = red[0][d] + red[1][d] + red[2][d] + red[3][d];
  const float inv = 1.f / S;
  half_t* ko = kT + ((size_t)bh * DH + d) * NTOK;
  for (int n = cg; n < NTOK; n += 4)
    ko[n] = (half_t)(__expf((float)kbase[(size_t)n * FDIM] - M) * inv);
  // v: scale by 1/N and transpose to [e][n]
  const half_t* vbase = qkv + b * NTOK * (size_t)FDIM + 2 * CDIM + h * DH;
  for (int idx = threadIdx.x; idx < DH * NTOK; idx += 256){
    const int e = idx >> 10, n = idx & 1023;
    vT[((size_t)bh * DH + e) * NTOK + n] =
        (half_t)((float)vbase[(size_t)n * FDIM + e] * (1.f / NTOK));
  }
}

// ---------------- Kernel 5: context = K^T V per (b,h); store context^T [e][d] ------
__global__ __launch_bounds__(128) void k_ctx(const half_t* __restrict__ kT,
                                             const half_t* __restrict__ vT,
                                             half_t* __restrict__ ctxT){
  const int bh = blockIdx.x;
  const int lane = threadIdx.x & 31, wid = threadIdx.x >> 5;
  const int l16 = lane & 15, hi = lane >> 4;
  const int khalf = hi * 8, kb = hi * 16;
  const int dbase = (wid >> 1) * 32, ebase = (wid & 1) * 32;
  v8f c00 = {}, c01 = {}, c10 = {}, c11 = {};
  const half_t* a0p = kT + ((size_t)bh * DH + dbase + l16) * NTOK;
  const half_t* a1p = a0p + 16 * NTOK;
  const half_t* b0p = vT + ((size_t)bh * DH + ebase + l16) * NTOK;
  const half_t* b1p = b0p + 16 * NTOK;
  for (int n0 = 0; n0 < NTOK; n0 += 32){
    v16h a0 = load_a_frag(a0p + n0, khalf);
    v16h a1 = load_a_frag(a1p + n0, khalf);
    v16h b0 = load_b_frag(b0p + n0, kb);
    v16h b1 = load_b_frag(b1p + n0, kb);
    c00 = WMMA(a0, b0, c00); c01 = WMMA(a0, b1, c01);
    c10 = WMMA(a1, b0, c10); c11 = WMMA(a1, b1, c11);
  }
  half_t* o = ctxT + (size_t)bh * DH * DH;
#pragma unroll
  for (int v = 0; v < 8; ++v){
    const int d0 = dbase + hi * 8 + v, d1 = d0 + 16;
    const int e0 = ebase + l16,        e1 = e0 + 16;
    o[e0 * DH + d0] = (half_t)c00[v];
    o[e1 * DH + d0] = (half_t)c01[v];
    o[e0 * DH + d1] = (half_t)c10[v];
    o[e1 * DH + d1] = (half_t)c11[v];
  }
}

// ---------------- Kernel 6: attn = Q x context; write [row, 512] f16 ----------------
__global__ __launch_bounds__(256) void k_attn(const half_t* __restrict__ qT,
                                              const half_t* __restrict__ ctxT,
                                              half_t* __restrict__ attn){
  const int bh = blockIdx.y;
  const size_t b = (size_t)(bh >> 3);
  const int h = bh & 7;
  const int lane = threadIdx.x & 31, wid = threadIdx.x >> 5;
  const int l16 = lane & 15, hi = lane >> 4;
  const int khalf = hi * 8, kb = hi * 16;
  const int mw = wid >> 1, nw = wid & 1;
  const int nbase = blockIdx.x * 128 + mw * 32;
  const int ebase = nw * 32;
  v8f c00 = {}, c01 = {}, c10 = {}, c11 = {};
  const half_t* a0p = qT + ((size_t)bh * NTOK + nbase + l16) * DH;
  const half_t* a1p = a0p + 16 * DH;
  const half_t* b0p = ctxT + (size_t)bh * DH * DH + (ebase + l16) * DH;
  const half_t* b1p = b0p + 16 * DH;
#pragma unroll
  for (int d0 = 0; d0 < DH; d0 += 32){
    v16h a0 = load_a_frag(a0p + d0, khalf);
    v16h a1 = load_a_frag(a1p + d0, khalf);
    v16h b0 = load_b_frag(b0p + d0, kb);
    v16h b1 = load_b_frag(b1p + d0, kb);
    c00 = WMMA(a0, b0, c00); c01 = WMMA(a0, b1, c01);
    c10 = WMMA(a1, b0, c10); c11 = WMMA(a1, b1, c11);
  }
#pragma unroll
  for (int v = 0; v < 8; ++v){
    const size_t n0 = (size_t)nbase + hi * 8 + v, n1 = n0 + 16;
    const int e0 = h * DH + ebase + l16, e1 = e0 + 16;
    attn[(b * NTOK + n0) * CDIM + e0] = (half_t)c00[v];
    attn[(b * NTOK + n0) * CDIM + e1] = (half_t)c01[v];
    attn[(b * NTOK + n1) * CDIM + e0] = (half_t)c10[v];
    attn[(b * NTOK + n1) * CDIM + e1] = (half_t)c11[v];
  }
}

// ---------------- Kernel 8: bias + LayerNorm2 + residual -> out --------------------
__global__ __launch_bounds__(256) void k_ln2(const float* __restrict__ ypre,
                                             const float* __restrict__ bias,
                                             const float* __restrict__ g,
                                             const float* __restrict__ x,
                                             float* __restrict__ out){
  const int lane = threadIdx.x & 31, wid = threadIdx.x >> 5;
  const size_t row = (size_t)blockIdx.x * 8 + wid;
  const size_t base = row * CDIM + lane * 16;
  float v[16]; float s = 0.f, sq = 0.f;
#pragma unroll
  for (int i = 0; i < 16; i += 4){
    float4 t  = *(const float4*)(ypre + base + i);
    float4 bb = *(const float4*)(bias + lane * 16 + i);
    v[i] = t.x + bb.x; v[i+1] = t.y + bb.y; v[i+2] = t.z + bb.z; v[i+3] = t.w + bb.w;
  }
#pragma unroll
  for (int i = 0; i < 16; i++){ s += v[i]; sq += v[i]*v[i]; }
  s = wsum(s); sq = wsum(sq);
  const float mu = s * (1.f/CDIM);
  const float var = sq * (1.f/CDIM) - mu*mu;
  const float rs = rsqrtf(var + 1e-5f);
#pragma unroll
  for (int i = 0; i < 16; i += 4){
    float4 gg = *(const float4*)(g + lane * 16 + i);
    float4 xx = *(const float4*)(x + base + i);
    float4 o;
    o.x = (v[i]   - mu) * rs * gg.x + xx.x;
    o.y = (v[i+1] - mu) * rs * gg.y + xx.y;
    o.z = (v[i+2] - mu) * rs * gg.z + xx.z;
    o.w = (v[i+3] - mu) * rs * gg.w + xx.w;
    *(float4*)(out + base + i) = o;
  }
}

extern "C" void kernel_launch(void* const* d_in, const int* in_sizes, int n_in,
                              void* d_out, int out_size, void* d_ws, size_t ws_size,
                              hipStream_t stream) {
  (void)in_sizes; (void)n_in; (void)out_size; (void)ws_size;
  const float* x    = (const float*)d_in[0];
  const float* ln1  = (const float*)d_in[1];
  const float* wqkv = (const float*)d_in[2];
  const float* wout = (const float*)d_in[3];
  const float* bout = (const float*)d_in[4];
  const float* ln2  = (const float*)d_in[5];
  float* out = (float*)d_out;

  char* ws = (char*)d_ws;
  half_t* xn    = (half_t*)(ws);                     //  32 MB  [RTOT x 512]  f16
  half_t* qkv   = (half_t*)(ws + (32ull  << 20));    //  96 MB  [RTOT x 1536] f16
  half_t* qT    = (half_t*)(ws + (128ull << 20));    //  32 MB  [b,h,n,d]     f16
  half_t* kT    = (half_t*)(ws + (160ull << 20));    //  32 MB  [b,h,d,n]     f16
  half_t* vT    = (half_t*)(ws + (192ull << 20));    //  32 MB  [b,h,e,n]     f16
  half_t* ctxT  = (half_t*)(ws + (224ull << 20));    //   2 MB  [b,h,e,d]     f16
  half_t* wqkvT = (half_t*)(ws + (228ull << 20));    // 1.5 MB  [1536 x 512]  f16
  half_t* woutT = (half_t*)(ws + (230ull << 20));    // 0.5 MB  [512 x 512]   f16
  half_t* attn  = (half_t*)(ws + (232ull << 20));    //  32 MB  [RTOT x 512]  f16
  float*  ypre  = (float*) (ws + (32ull  << 20));    //  64 MB, aliases dead qkv

  k_wprep<<<FDIM, 256, 0, stream>>>(wqkv, FDIM, wqkvT);
  k_wprep<<<CDIM, 256, 0, stream>>>(wout, CDIM, woutT);
  k_ln1<<<RTOT/8, 256, 0, stream>>>(x, ln1, xn);
  k_gemm<true ><<<dim3(FDIM/128, RTOT/64), 256, 0, stream>>>(xn, wqkvT, FDIM, (void*)qkv);
  k_qsm<<<RTOT, 256, 0, stream>>>(qkv, qT);                 // RTOT*8 pairs / 8 per block
  k_ksm_v<<<32*NH, 256, 0, stream>>>(qkv, kT, vT);
  k_ctx<<<32*NH, 128, 0, stream>>>(kT, vT, ctxT);
  k_attn<<<dim3(NTOK/128, 32*NH), 256, 0, stream>>>(qT, ctxT, attn);
  k_gemm<false><<<dim3(CDIM/128, RTOT/64), 256, 0, stream>>>(attn, woutT, CDIM, (void*)ypre);
  k_ln2<<<RTOT/8, 256, 0, stream>>>(ypre, bout, ln2, x, out);
}
